// SpikingResidualTemporalBlock_11484742550219
// MI455X (gfx1250) — matrix operand order repaired and checked by reading, not compiled, not run on hardware
//
#include <hip/hip_runtime.h>
#include <hip/hip_bf16.h>

// ---------------------------------------------------------------------------
// SpikingResidualTemporalBlock for MI455X (gfx1250, wave32, WMMA + TDM)
//   x:[B=32,T=512,D=512] -> fc1(H=1024)+LN -> LIF -> fc2(D=512)+LN -> LIF -> +x
// GEMMs: bf16 WMMA (v_wmma_f32_16x16x32_bf16), A-stripe staged to LDS via the
// Tensor Data Mover (tensor_load_to_lds + s_wait_tensorcnt), B double-buffered
// in registers so WMMAs overlap global loads.
// ---------------------------------------------------------------------------

typedef __attribute__((ext_vector_type(8)))  __bf16        v8bf;
typedef __attribute__((ext_vector_type(16))) __bf16        v16bf;
typedef __attribute__((ext_vector_type(8)))  float         v8f;
typedef __attribute__((ext_vector_type(4)))  unsigned int  v4u;
typedef __attribute__((ext_vector_type(8)))  int           v8i;
typedef __attribute__((ext_vector_type(4)))  int           v4i;

#define B_DIM 32
#define T_DIM 512
#define D_DIM 512
#define H_DIM 1024
#define BT    (B_DIM * T_DIM)   // 16384 rows

#if defined(__HIP_DEVICE_COMPILE__) && __has_builtin(__builtin_amdgcn_tensor_load_to_lds) && __has_builtin(__builtin_amdgcn_s_wait_tensorcnt)
#define HAVE_TDM 1
#else
#define HAVE_TDM 0
#endif

// ---- fragment loaders (16-bit WMMA operand layouts, wave32) ----------------
// A-matrix 16x32 bf16 (from LDS stripe): lane<16 -> row=lane,  K=[0..7]+[16..23]
//                                        lane>=16-> row=l-16,  K=[8..15]+[24..31]
__device__ __forceinline__ v16bf load_a_frag_lds(const __bf16* __restrict__ base,
                                                 int ldk, int k0, int lane) {
    const int m  = lane & 15;
    const int hf = lane >> 4;
    const __bf16* p = base + (size_t)m * ldk + k0 + 8 * hf;
    v8bf lo = *reinterpret_cast<const v8bf*>(p);        // K = k0+8*hf    .. +8
    v8bf hi = *reinterpret_cast<const v8bf*>(p + 16);   // K = k0+16+8*hf .. +8
    return __builtin_shufflevector(lo, hi, 0,1,2,3,4,5,6,7,8,9,10,11,12,13,14,15);
}

// B-matrix 32x16 bf16 (K x N); W stored N rows x K cols (K-major):
// lane<16 -> col=lane, K=[0..7]+[16..23];  lane>=16 -> col=lane-16, K=[8..15]+[24..31]
__device__ __forceinline__ v16bf load_b_frag(const __bf16* __restrict__ W,
                                             int ldk, int lane) {
    const int n  = lane & 15;
    const int hf = lane >> 4;
    const __bf16* p = W + (size_t)n * ldk + 8 * hf;
    v8bf lo = *reinterpret_cast<const v8bf*>(p);
    v8bf hi = *reinterpret_cast<const v8bf*>(p + 16);
    return __builtin_shufflevector(lo, hi, 0,1,2,3,4,5,6,7,8,9,10,11,12,13,14,15);
}

template <int PW>
__device__ __forceinline__ void load_b_set(const __bf16* __restrict__ W, int ldk,
                                           int k0, int colBase, int lane,
                                           v16bf (&bf)[PW]) {
    #pragma unroll
    for (int i = 0; i < PW; ++i)
        bf[i] = load_b_frag(W + (size_t)(colBase + i * 16) * ldk + k0, ldk, lane);
}

// ---- fp32 -> bf16 conversion ----------------------------------------------
__global__ void f32_to_bf16_kernel(const float* __restrict__ in,
                                   __bf16* __restrict__ out, int n) {
    int i = blockIdx.x * blockDim.x + threadIdx.x;
    if (i < n) out[i] = (__bf16)in[i];
}

// ---- fused GEMM (A[BT,K] * W[N,K]^T) + bias + LayerNorm --------------------
// One workgroup owns a 16-row x NCOLS stripe (LN is workgroup-local).
// PW = 4 WMMA N-tiles per wave; THREADS = (NCOLS/64)*32  (512 or 256).
template <int NCOLS, int KDIM>
__global__ void __launch_bounds__((NCOLS / 64) * 32)
gemm_bias_ln_kernel(const __bf16* __restrict__ A,
                    const __bf16* __restrict__ W,
                    const float*  __restrict__ bias,
                    const float*  __restrict__ gamma,
                    const float*  __restrict__ beta,
                    float*        __restrict__ out) {
    constexpr int PW      = 4;
    constexpr int THREADS = (NCOLS / 64) * 32;
    constexpr int SUBS    = THREADS / 16;          // threads per row in LN

    __shared__ __bf16 aStripe[16 * KDIM];          // 16 or 32 KB
    __shared__ float  tile[16 * NCOLS];            // 64 or 32 KB
    __shared__ float  redS[THREADS];
    __shared__ float  redQ[THREADS];
    __shared__ float  muA[16];
    __shared__ float  rsA[16];

    const int tid  = threadIdx.x;
    const int lane = tid & 31;
    const int wave = tid >> 5;
    const int hf   = lane >> 4;
    const int idx  = lane & 15;
    const int row0 = blockIdx.x * 16;
    const int colBase = wave * (PW * 16);

    // ---- stage the 16 x KDIM A stripe into LDS --------------------------------
#if HAVE_TDM
    if (tid < 32) {   // single wave issues the TDM op (TDM ignores EXEC)
        const unsigned long long ga =
            (unsigned long long)(const void*)(A + (size_t)row0 * KDIM);
        const unsigned int lds_off =
            (unsigned int)(unsigned long long)(const void*)aStripe;
        v4u g0;
        g0[0] = 1u;                                          // count=1 (user D#)
        g0[1] = lds_off;                                     // lds_addr
        g0[2] = (unsigned int)ga;                            // global_addr[31:0]
        g0[3] = (unsigned int)((ga >> 32) & 0x1FFFFFFull)    // global_addr[56:32]
              | (2u << 30);                                  // type = 2 (image)
        v8i g1;
        g1[0] = (int)(1u << 16);                             // data_size = 2B
        g1[1] = (int)(((unsigned)KDIM & 0xFFFFu) << 16);     // tensor_dim0 lo
        g1[2] = (int)(((unsigned)KDIM >> 16) |
                      ((unsigned)(BT & 0xFFFF) << 16));      // dim0 hi | tensor_dim1 lo
        g1[3] = (int)(((unsigned)BT >> 16) |
                      ((unsigned)KDIM << 16));               // dim1 hi | tile_dim0 = KDIM
        g1[4] = 16;                                          // tile_dim1 = 16 rows
        g1[5] = (int)KDIM;                                   // tensor_dim0_stride lo
        g1[6] = 0;
        g1[7] = 0;
        v4i gz4 = {0, 0, 0, 0};                              // groups 2/3: 2D tensor
        v8i gz8 = {0, 0, 0, 0, 0, 0, 0, 0};                  // trailing group (6-arg form)
        __builtin_amdgcn_tensor_load_to_lds(g0, g1, gz4, gz4, gz8, 0);
        __builtin_amdgcn_s_wait_tensorcnt(0);
    }
#else
    {
        const v8bf* gsrc = reinterpret_cast<const v8bf*>(A + (size_t)row0 * KDIM);
        v8bf*       ldst = reinterpret_cast<v8bf*>(aStripe);
        for (int c = tid; c < 2 * KDIM; c += THREADS) ldst[c] = gsrc[c];
    }
#endif
    __syncthreads();

    // ---- main loop: double-buffered B fragments, A from LDS -------------------
    v8f acc[PW];
    #pragma unroll
    for (int i = 0; i < PW; ++i) acc[i] = (v8f){0.f,0.f,0.f,0.f,0.f,0.f,0.f,0.f};

    v16bf b0[PW], b1[PW];
    load_b_set<PW>(W, KDIM, 0, colBase, lane, b0);

    for (int k0 = 0; k0 < KDIM; k0 += 64) {
        load_b_set<PW>(W, KDIM, k0 + 32, colBase, lane, b1);     // prefetch k+32
        v16bf a0 = load_a_frag_lds(aStripe, KDIM, k0, lane);
        #pragma unroll
        for (int i = 0; i < PW; ++i)
            acc[i] = __builtin_amdgcn_wmma_f32_16x16x32_bf16(
                         false, a0, false, b0[i], (short)0, acc[i], false, false);

        if (k0 + 64 < KDIM)
            load_b_set<PW>(W, KDIM, k0 + 64, colBase, lane, b0); // prefetch k+64
        v16bf a1 = load_a_frag_lds(aStripe, KDIM, k0 + 32, lane);
        #pragma unroll
        for (int i = 0; i < PW; ++i)
            acc[i] = __builtin_amdgcn_wmma_f32_16x16x32_bf16(
                         false, a1, false, b1[i], (short)0, acc[i], false, false);
    }

    // ---- epilogue: + bias, stash stripe to LDS --------------------------------
    // C/D layout: acc[r] of lane(hf,idx) is element (row = r + 8*hf, col = idx)
    #pragma unroll
    for (int i = 0; i < PW; ++i) {
        const int col = colBase + i * 16 + idx;
        const float bv = bias[col];
        #pragma unroll
        for (int r = 0; r < 8; ++r)
            tile[(r + 8 * hf) * NCOLS + col] = acc[i][r] + bv;
    }
    __syncthreads();

    // ---- LayerNorm over NCOLS: SUBS threads per row ---------------------------
    const int row = tid / SUBS;
    const int sub = tid % SUBS;
    float s = 0.f, q = 0.f;
    for (int c = sub; c < NCOLS; c += SUBS) {
        float v = tile[row * NCOLS + c];
        s += v; q += v * v;
    }
    redS[tid] = s; redQ[tid] = q;
    __syncthreads();
    if (sub == 0) {
        float S = 0.f, Q = 0.f;
        #pragma unroll
        for (int i = 0; i < SUBS; ++i) { S += redS[row * SUBS + i]; Q += redQ[row * SUBS + i]; }
        const float m   = S * (1.0f / NCOLS);
        const float var = Q * (1.0f / NCOLS) - m * m;
        muA[row] = m;
        rsA[row] = rsqrtf(var + 1e-5f);
    }
    __syncthreads();

    for (int j = tid; j < 16 * NCOLS; j += THREADS) {
        const int r = j / NCOLS;
        const int c = j % NCOLS;
        out[(size_t)(row0 + r) * NCOLS + c] =
            (tile[j] - muA[r]) * rsA[r] * gamma[c] + beta[c];
    }
}

// ---- LIF multistep (scan over T), emit bf16 spikes -------------------------
// h layout [B,T,C]; one thread per (b,c); v' = 0.5*(v + x_t); spike v>=1; hard reset.
__global__ void lif_spike_kernel(const float* __restrict__ h,
                                 __bf16* __restrict__ spk, int C) {
    const int id = blockIdx.x * blockDim.x + threadIdx.x;   // over B*C
    const int b  = id / C;
    const int ch = id % C;
    size_t off = (size_t)b * T_DIM * C + ch;
    float v = 0.f;
    #pragma unroll 4
    for (int t = 0; t < T_DIM; ++t) {
        const float xv = h[off];
        v = v + (xv - v) * 0.5f;                 // tau = 2
        const float sp = (v >= 1.0f) ? 1.f : 0.f;
        spk[off] = (__bf16)sp;                    // {0,1} exact in bf16
        v = v * (1.f - sp);
        off += (size_t)C;
    }
}

// ---- LIF multistep fused with residual add, fp32 output --------------------
__global__ void lif_residual_kernel(const float* __restrict__ y,
                                    const float* __restrict__ x,
                                    float* __restrict__ out, int C) {
    const int id = blockIdx.x * blockDim.x + threadIdx.x;   // over B*C
    const int b  = id / C;
    const int ch = id % C;
    size_t off = (size_t)b * T_DIM * C + ch;
    float v = 0.f;
    #pragma unroll 4
    for (int t = 0; t < T_DIM; ++t) {
        const float yv = y[off];
        v = v + (yv - v) * 0.5f;
        const float sp = (v >= 1.0f) ? 1.f : 0.f;
        out[off] = x[off] + sp;                   // residual + spike
        v = v * (1.f - sp);
        off += (size_t)C;
    }
}

// ---------------------------------------------------------------------------
static inline size_t align256(size_t x) { return (x + 255) & ~(size_t)255; }

extern "C" void kernel_launch(void* const* d_in, const int* in_sizes, int n_in,
                              void* d_out, int out_size, void* d_ws, size_t ws_size,
                              hipStream_t stream) {
    (void)in_sizes; (void)n_in; (void)out_size; (void)ws_size;

    const float* x   = (const float*)d_in[0];   // [B,T,D]
    const float* w1  = (const float*)d_in[1];   // [H,D]
    const float* b1  = (const float*)d_in[2];   // [H]
    const float* g1  = (const float*)d_in[3];   // [H]
    const float* be1 = (const float*)d_in[4];   // [H]
    const float* w2  = (const float*)d_in[5];   // [D,H]
    const float* b2  = (const float*)d_in[6];   // [D]
    const float* g2  = (const float*)d_in[7];   // [D]
    const float* be2 = (const float*)d_in[8];   // [D]
    float* out = (float*)d_out;                 // [B,T,D]

    // workspace carve-up
    char* ws = (char*)d_ws;
    __bf16* xbf  = (__bf16*)ws;  ws += align256((size_t)BT * D_DIM * sizeof(__bf16));
    __bf16* w1bf = (__bf16*)ws;  ws += align256((size_t)H_DIM * D_DIM * sizeof(__bf16));
    __bf16* w2bf = (__bf16*)ws;  ws += align256((size_t)D_DIM * H_DIM * sizeof(__bf16));
    float*  hln  = (float*) ws;  ws += align256((size_t)BT * H_DIM * sizeof(float));
    __bf16* spk  = (__bf16*)ws;  ws += align256((size_t)BT * H_DIM * sizeof(__bf16));
    float*  yln  = (float*) ws;  ws += align256((size_t)BT * D_DIM * sizeof(float));

    // 1) convert x, w1, w2 to bf16
    {
        const int n_x = BT * D_DIM;
        f32_to_bf16_kernel<<<(n_x + 255) / 256, 256, 0, stream>>>(x, xbf, n_x);
        const int n_w1 = H_DIM * D_DIM;
        f32_to_bf16_kernel<<<(n_w1 + 255) / 256, 256, 0, stream>>>(w1, w1bf, n_w1);
        const int n_w2 = D_DIM * H_DIM;
        f32_to_bf16_kernel<<<(n_w2 + 255) / 256, 256, 0, stream>>>(w2, w2bf, n_w2);
    }

    // 2) fc1 + bias + LN1  (A = xbf [BT,512], W = w1bf [1024,512]) : 512 thr
    gemm_bias_ln_kernel<H_DIM, D_DIM><<<BT / 16, (H_DIM / 64) * 32, 0, stream>>>(
        xbf, w1bf, b1, g1, be1, hln);

    // 3) LIF over T -> bf16 spikes [B,T,H]
    lif_spike_kernel<<<(B_DIM * H_DIM) / 256, 256, 0, stream>>>(hln, spk, H_DIM);

    // 4) fc2 + bias + LN2  (A = spk [BT,1024], W = w2bf [512,1024]) : 256 thr
    gemm_bias_ln_kernel<D_DIM, H_DIM><<<BT / 16, (D_DIM / 64) * 32, 0, stream>>>(
        spk, w2bf, b2, g2, be2, yln);

    // 5) LIF over T + residual -> out fp32 [B,T,D]
    lif_residual_kernel<<<(B_DIM * D_DIM) / 256, 256, 0, stream>>>(yln, x, out, D_DIM);
}
